// SAE_topk_31370441130588
// MI455X (gfx1250) — matrix operand order repaired and checked by reading, not compiled, not run on hardware
//
#include <hip/hip_runtime.h>

typedef __attribute__((ext_vector_type(16))) __bf16 bf16x16;
typedef __attribute__((ext_vector_type(8)))  __bf16 bf16x8;
typedef __attribute__((ext_vector_type(4)))  __bf16 bf16x4;
typedef __attribute__((ext_vector_type(8)))  float  f32x8;
typedef __attribute__((ext_vector_type(4)))  float  f32x4;

#define TOKENS  8192
#define INPUT   1024
#define HIDDEN  16384
#define KSEL    32

#define M_TOK     32            // tokens per block (one WMMA M=16 pair)
#define NWAVE     8
#define NTHREADS  256
#define N_TILE    (NWAVE * 16)  // 128 hidden columns per iteration

// CDNA5 async global->LDS copy (16B per enabled lane), tracked by ASYNCcnt.
__device__ __forceinline__ void async_g2lds_b128(const void* gsrc, unsigned lds_off) {
  asm volatile("global_load_async_to_lds_b128 %0, %1, off"
               :: "v"(lds_off), "v"(gsrc) : "memory");
}
__device__ __forceinline__ void wait_asynccnt0() {
  asm volatile("s_wait_asynccnt 0x0" ::: "memory");
}

// ---------------------------------------------------------------------------
// Prep 1: xcb[t][k] = bf16(x[t][k] - pre_encode_b[k])
// ---------------------------------------------------------------------------
__global__ __launch_bounds__(256)
void prep_x_kernel(const float* __restrict__ x, const float* __restrict__ pb,
                   __bf16* __restrict__ xcb) {
  int e = (blockIdx.x * 256 + threadIdx.x) * 4;
  int k = e & (INPUT - 1);
  f32x4 xv = *(const f32x4*)(x + e);
  f32x4 bv = *(const f32x4*)(pb + k);
  bf16x4 o;
#pragma unroll
  for (int i = 0; i < 4; ++i) o[i] = (__bf16)(xv[i] - bv[i]);
  *(bf16x4*)(xcb + e) = o;
}

// ---------------------------------------------------------------------------
// Prep 2: wtb[n][k] = bf16(WT[k][n])   (transpose so K is contiguous per column)
// ---------------------------------------------------------------------------
__global__ __launch_bounds__(256)
void prep_wt_kernel(const float* __restrict__ WT, __bf16* __restrict__ wtb) {
  __shared__ float tile[32][33];
  int n0 = blockIdx.x * 32;          // hidden
  int k0 = blockIdx.y * 32;          // input
  int tx = threadIdx.x & 31, ty = threadIdx.x >> 5;   // ty in 0..7
#pragma unroll
  for (int i = 0; i < 32; i += 8)
    tile[ty + i][tx] = WT[(size_t)(k0 + ty + i) * HIDDEN + n0 + tx];
  __syncthreads();
#pragma unroll
  for (int i = 0; i < 32; i += 8)
    wtb[(size_t)(n0 + ty + i) * INPUT + k0 + tx] = (__bf16)tile[tx][ty + i];
}

// ---------------------------------------------------------------------------
// Fused encode GEMM (bf16 WMMA, f32 accum) + streaming per-token top-32.
// Block: 32 tokens, 8 waves. Wave w owns cols [hb+16w, hb+16w+16).
// Lane l maintains top-32 of (token=l, wave-slice=w) in LDS; merged at end.
// ---------------------------------------------------------------------------
__global__ __launch_bounds__(NTHREADS)
void encode_topk_kernel(const __bf16* __restrict__ xcb,
                        const __bf16* __restrict__ wtb,
                        const float* __restrict__ b1,
                        float* __restrict__ tv, int* __restrict__ ti) {
  extern __shared__ char smem[];
  __bf16* Ab  = (__bf16*)smem;                            // 32*1024*2 = 65536 B
  float*  stg = (float*)(smem + 65536);                   // 8*32*16*4 = 16384 B
  float*  kv  = (float*)(smem + 65536 + 16384);           // 8*32*32*4 = 32768 B
  int*    ki  = (int*)  (smem + 65536 + 16384 + 32768);   // 32768 B   (144 KB tot)

  const int tid  = threadIdx.x;
  const int lane = tid & 31, w = tid >> 5;
  const int half = lane >> 4, nin = lane & 15;
  const int tok0 = blockIdx.x * M_TOK;

  // A tile (32 tokens x 1024 K, bf16) -> LDS via async DMA, no VGPR round-trip
  {
    const bf16x8* src = (const bf16x8*)(xcb + (size_t)tok0 * INPUT);
    bf16x8* dst = (bf16x8*)Ab;
#pragma unroll
    for (int i = tid; i < M_TOK * INPUT / 8; i += NTHREADS)
      async_g2lds_b128(src + i, (unsigned)(size_t)(dst + i));
  }
  for (int i = tid; i < NWAVE * M_TOK * KSEL; i += NTHREADS) kv[i] = -INFINITY;
  wait_asynccnt0();
  __syncthreads();

  // per-lane running min of my (token=lane, slice=w) list
  float mv = -INFINITY; int mp = 0;
  float* mykv = kv + (w * M_TOK + lane) * KSEL;
  int*   myki = ki + (w * M_TOK + lane) * KSEL;

  // A-operand LDS element offset: row (lane&15), K chunk select by lane half
  const int aoff = (lane & 15) * INPUT + half * 16;

  for (int hb = 0; hb < HIDDEN; hb += N_TILE) {
    const int ncol = hb + w * 16 + nin;
    const __bf16* Bb = wtb + (size_t)ncol * INPUT + half * 16;
    if (hb + N_TILE < HIDDEN) {
      // hide L2 latency for the next hidden tile's B stream
      __builtin_prefetch(Bb + (size_t)N_TILE * INPUT, 0, 0);
    }
    f32x8 acc0 = {}; f32x8 acc1 = {};
#pragma unroll 4
    for (int kc = 0; kc < INPUT / 32; ++kc) {
      bf16x16 b = *(const bf16x16*)(Bb + kc * 32);              // 32B global (L2)
      bf16x8 a0l = *(const bf16x8*)(Ab + aoff + kc * 32);
      bf16x8 a0h = *(const bf16x8*)(Ab + aoff + kc * 32 + 8);
      bf16x8 a1l = *(const bf16x8*)(Ab + aoff + 16 * INPUT + kc * 32);
      bf16x8 a1h = *(const bf16x8*)(Ab + aoff + 16 * INPUT + kc * 32 + 8);
      bf16x16 a0, a1;
#pragma unroll
      for (int i = 0; i < 8; ++i) {
        a0[i] = a0l[i]; a0[i + 8] = a0h[i];
        a1[i] = a1l[i]; a1[i + 8] = a1h[i];
      }
      acc0 = __builtin_amdgcn_wmma_f32_16x16x32_bf16(false, a0, false, b,
                                                     (short)0, acc0, false, false);
      acc1 = __builtin_amdgcn_wmma_f32_16x16x32_bf16(false, a1, false, b,
                                                     (short)0, acc1, false, false);
    }

    const float b1v = b1[ncol];
    // stage D tiles: element (m, nin) of acc_r lives in VGPR i, lane half
    float* s = stg + w * (M_TOK * 16);
#pragma unroll
    for (int i = 0; i < 8; ++i) {
      s[(half * 8 + i) * 16 + nin]      = acc0[i] + b1v;
      s[(16 + half * 8 + i) * 16 + nin] = acc1[i] + b1v;
    }
    __syncthreads();

    // lane scans its token's 16 new values, threshold-filtered insert
    const float* row = stg + w * (M_TOK * 16) + lane * 16;
#pragma unroll
    for (int c = 0; c < 16; ++c) {
      float v = row[c];
      if (v > mv) {
        mykv[mp] = v; myki[mp] = hb + w * 16 + c;
        float nm = INFINITY; int np = 0;
#pragma unroll
        for (int e = 0; e < KSEL; ++e) {
          float q = mykv[e];
          if (q < nm) { nm = q; np = e; }
        }
        mv = nm; mp = np;
      }
    }
    __syncthreads();
  }

  // merge the 8 per-wave partial lists per token (order-free top-32)
  if (tid < M_TOK) {
    float* fkv = kv + tid * KSEL;      // slice 0 becomes the running list
    int*   fki = ki + tid * KSEL;
    float nm = INFINITY; int np = 0;
    for (int e = 0; e < KSEL; ++e) { float q = fkv[e]; if (q < nm) { nm = q; np = e; } }
    for (int ww = 1; ww < NWAVE; ++ww) {
      const float* sv = kv + (ww * M_TOK + tid) * KSEL;
      const int*   si = ki + (ww * M_TOK + tid) * KSEL;
      for (int e = 0; e < KSEL; ++e) {
        float v = sv[e];
        if (v > nm) {
          fkv[np] = v; fki[np] = si[e];
          nm = INFINITY;
          for (int e2 = 0; e2 < KSEL; ++e2) {
            float q = fkv[e2];
            if (q < nm) { nm = q; np = e2; }
          }
        }
      }
    }
    const size_t t = (size_t)(tok0 + tid);
    for (int e = 0; e < KSEL; ++e) { tv[t * KSEL + e] = fkv[e]; ti[t * KSEL + e] = fki[e]; }
  }
}

// ---------------------------------------------------------------------------
// Decode: x_hat[t,:] = sum_k vals[t,k] * W[idx[t,k],:] + b2   (full fp32)
// ---------------------------------------------------------------------------
__global__ __launch_bounds__(256)
void decode_kernel(const float* __restrict__ tv, const int* __restrict__ ti,
                   const float* __restrict__ W, const float* __restrict__ b2,
                   float* __restrict__ out) {
  const int t = blockIdx.x;
  const int c = threadIdx.x;
  float a0 = b2[c], a1 = b2[c + 256], a2 = b2[c + 512], a3 = b2[c + 768];
  const float* v  = tv + (size_t)t * KSEL;
  const int*   id = ti + (size_t)t * KSEL;
#pragma unroll 4
  for (int kk = 0; kk < KSEL; ++kk) {
    float vv = v[kk];
    const float* row = W + (size_t)id[kk] * INPUT;
    a0 += vv * row[c];
    a1 += vv * row[c + 256];
    a2 += vv * row[c + 512];
    a3 += vv * row[c + 768];
  }
  float* o = out + (size_t)t * INPUT;
  o[c] = a0; o[c + 256] = a1; o[c + 512] = a2; o[c + 768] = a3;
}

// ---------------------------------------------------------------------------
extern "C" void kernel_launch(void* const* d_in, const int* in_sizes, int n_in,
                              void* d_out, int out_size, void* d_ws, size_t ws_size,
                              hipStream_t stream) {
  const float* x  = (const float*)d_in[0];
  const float* pb = (const float*)d_in[1];
  const float* W  = (const float*)d_in[2];
  const float* WT = (const float*)d_in[3];
  const float* b1 = (const float*)d_in[4];
  const float* b2 = (const float*)d_in[5];
  float* out = (float*)d_out;

  char* ws = (char*)d_ws;
  __bf16* xcb = (__bf16*)ws;                               // 16 MB
  __bf16* wtb = (__bf16*)(ws + ((size_t)16 << 20));        // 32 MB (L2-resident B)
  float*  tv  = (float*) (ws + ((size_t)48 << 20));        //  1 MB
  int*    ti  = (int*)   (ws + ((size_t)49 << 20));        //  1 MB

  prep_x_kernel<<<TOKENS * INPUT / (256 * 4), 256, 0, stream>>>(x, pb, xcb);
  dim3 tg(HIDDEN / 32, INPUT / 32);
  prep_wt_kernel<<<tg, 256, 0, stream>>>(WT, wtb);

  const size_t smem = 65536 + 16384 + 32768 + 32768;       // 144 KB dynamic LDS
  encode_topk_kernel<<<TOKENS / M_TOK, NTHREADS, smem, stream>>>(xcb, wtb, b1, tv, ti);

  decode_kernel<<<TOKENS, 256, 0, stream>>>(tv, ti, W, b2, out);
}
